// SMILESDecoder_40647570489355
// MI455X (gfx1250) — compile-verified, matrix operand order checked by reading, not run on hardware
//
#include <hip/hip_runtime.h>
#include <hip/hip_bf16.h>

// ---------------------------------------------------------------------------
// SMILES transformer decoder forward, CDNA5 (gfx1250), wave32 WMMA bf16 path.
// ---------------------------------------------------------------------------

#define Bv   64
#define Sv   256
#define Dv   512
#define Hv   8
#define DFFv 2048
#define Vv   128
#define Lv   6
#define NEGV (-1000000000.0f)

typedef __attribute__((ext_vector_type(16))) __bf16 v16bf;
typedef __attribute__((ext_vector_type(8)))  float  v8f;

union FragU { v16bf v; uint4 q[2]; };

// ---- CDNA5 async global->LDS copy (zero data VGPRs, ASYNCcnt tracked) ----
#if defined(__has_builtin)
#  if __has_builtin(__builtin_amdgcn_global_load_async_to_lds_b128) && \
      __has_builtin(__builtin_amdgcn_s_wait_asynccnt)
#    define USE_ASYNC_LDS 1
#  endif
#endif
#ifndef USE_ASYNC_LDS
#  define USE_ASYNC_LDS 0
#endif

#if USE_ASYNC_LDS
// The builtin expects pointers to int4 vectors (per hipcc diagnostic:
// "__attribute__((__vector_size__(4 * sizeof(int)))) int *").  Pass generic
// pointers with that exact pointee type; clang rewrites the builtin's pointer
// address spaces to match and codegen inserts the addrspace casts.
typedef int v4i_vs __attribute__((vector_size(16)));
__device__ __forceinline__ void cp_async16(const unsigned short* g,
                                           unsigned short* l) {
    __builtin_amdgcn_global_load_async_to_lds_b128(
        (v4i_vs*)(unsigned long long)g,   // strip const, keep generic AS
        (v4i_vs*)l,                       // generic LDS pointer (do not truncate)
        0, 0);
}
__device__ __forceinline__ void async_wait0() {
    __builtin_amdgcn_s_wait_asynccnt(0);
}
#else
__device__ __forceinline__ void async_wait0() {}
#endif

__device__ __forceinline__ unsigned short f2bf(float x) {
    union { float f; unsigned u; } c; c.f = x;
    unsigned r = c.u + 0x7FFFu + ((c.u >> 16) & 1u);
    return (unsigned short)(r >> 16);
}

// Load a 16-element bf16 fragment slice for one lane.
// Caller bakes in: base + (lane&15)*ld + (lane>>4)*8 (+ k offset).
// Loads bytes [0,16) and [32,48) => K pattern {0..7,16..23} / {8..15,24..31}.
__device__ __forceinline__ v16bf ld_frag(const unsigned short* p) {
    FragU f;
    f.q[0] = *(const uint4*)(p);
    f.q[1] = *(const uint4*)(p + 16);
    return f.v;
}

__device__ __forceinline__ v8f wmma_bf16(v16bf a, v16bf b, v8f c) {
    return __builtin_amdgcn_wmma_f32_16x16x32_bf16(
        false, a, false, b, (short)0, c, false, false);
}

// ---------------------------------------------------------------------------
// fp32 -> bf16 conversion (grid-stride)
// ---------------------------------------------------------------------------
__global__ void __launch_bounds__(256) cvt_bf16(const float* __restrict__ src,
                                                unsigned short* __restrict__ dst,
                                                int n) {
    for (int i = blockIdx.x * blockDim.x + threadIdx.x; i < n;
         i += gridDim.x * blockDim.x)
        dst[i] = f2bf(src[i]);
}

// ---------------------------------------------------------------------------
// Embedding + positional: x[b,s,:] = emb[tgt[b,s],:] + pos[s,:]
// ---------------------------------------------------------------------------
__global__ void __launch_bounds__(256) embed_kernel(const int* __restrict__ tgt,
                                                    const float* __restrict__ emb,
                                                    const float* __restrict__ pos,
                                                    float* __restrict__ xf,
                                                    unsigned short* __restrict__ xb) {
    const int total = Bv * Sv * Dv;
    for (int i = blockIdx.x * blockDim.x + threadIdx.x; i < total;
         i += gridDim.x * blockDim.x) {
        int row = i >> 9;         // /D
        int c   = i & (Dv - 1);
        int s   = row & (Sv - 1);
        int tok = tgt[row];
        float v = emb[tok * Dv + c] + pos[s * Dv + c];
        xf[i] = v;
        xb[i] = f2bf(v);
    }
}

// ---------------------------------------------------------------------------
// Generic NT GEMM: C[M,N] = A[M,K](bf16) * W[N,K](bf16)^T + bias
// Block: 128 threads (4 waves), tile 128x128, wave tile 64x64.
// Double-buffered LDS, one barrier per k-step. Panel staging uses CDNA5
// async global->LDS DMA when available (no data VGPRs => no spills); the
// fallback path stages *after* the WMMA block so staged values have a
// 2-instruction live range.
// flags: 1=relu, 2=store f32, 4=store bf16
// N must be a multiple of 128; K a multiple of 32; M guarded at store.
// ---------------------------------------------------------------------------
__global__ void __launch_bounds__(128)
gemm_nt(const unsigned short* __restrict__ A,
        const unsigned short* __restrict__ W,
        const float* __restrict__ bias,
        float* __restrict__ Cf,
        unsigned short* __restrict__ Cb,
        int M, int N, int K, int flags) {
    __shared__ unsigned short As[2][128 * 32];
    __shared__ unsigned short Ws[2][128 * 32];

    const int tid  = threadIdx.x;
    const int lane = tid & 31;
    const int wave = tid >> 5;
    const int wm   = wave >> 1, wn = wave & 1;
    const int lm   = lane & 15, half = lane >> 4;
    const int mt = blockIdx.y * 128, nt = blockIdx.x * 128;

    v8f acc[4][4];
#pragma unroll
    for (int i = 0; i < 4; i++)
#pragma unroll
        for (int j = 0; j < 4; j++) acc[i][j] = (v8f){0,0,0,0,0,0,0,0};

    // Clamp tile rows instead of zero-filling: A row m only feeds C row m,
    // so duplicated rows only produce values in C rows >= M / cols >= N,
    // which the epilogue never stores.
    const int aRow = min(mt + tid, M - 1);
    const int wRow = min(nt + tid, N - 1);
    const unsigned short* aPtr = A + (size_t)aRow * K;
    const unsigned short* wPtr = W + (size_t)wRow * K;

    // stage one 32-wide k-panel (both operands) into LDS buffer b2
    auto stage = [&](int b2, int kk2) {
#if USE_ASYNC_LDS
#pragma unroll
        for (int i = 0; i < 4; i++) {
            cp_async16(aPtr + kk2 + i * 8, As[b2] + tid * 32 + i * 8);
            cp_async16(wPtr + kk2 + i * 8, Ws[b2] + tid * 32 + i * 8);
        }
#else
#pragma unroll
        for (int i = 0; i < 4; i++) {
            uint4 a4 = *(const uint4*)(aPtr + kk2 + i * 8);
            uint4 w4 = *(const uint4*)(wPtr + kk2 + i * 8);
            *(uint4*)(As[b2] + tid * 32 + i * 8) = a4;
            *(uint4*)(Ws[b2] + tid * 32 + i * 8) = w4;
        }
#endif
    };

    stage(0, 0);
    async_wait0();
    __syncthreads();

    int buf = 0;
    for (int kk = 0; kk < K; kk += 32) {
        const bool more = (kk + 32) < K;
#if USE_ASYNC_LDS
        // async DMA for next panel overlaps the WMMA block below
        if (more) stage(buf ^ 1, kk + 32);
#endif
        if (kk + 64 < K) {
            __builtin_prefetch(aPtr + kk + 64, 0, 1);   // global_prefetch_b8
            __builtin_prefetch(wPtr + kk + 64, 0, 1);
        }

        const unsigned short* ab = As[buf] + (wm * 64 + lm) * 32 + half * 8;
        const unsigned short* wb = Ws[buf] + (wn * 64 + lm) * 32 + half * 8;
        v16bf af[4], wf[4];
#pragma unroll
        for (int i = 0; i < 4; i++) af[i] = ld_frag(ab + i * 16 * 32);
#pragma unroll
        for (int j = 0; j < 4; j++) wf[j] = ld_frag(wb + j * 16 * 32);
#pragma unroll
        for (int i = 0; i < 4; i++)
#pragma unroll
            for (int j = 0; j < 4; j++)
                acc[i][j] = wmma_bf16(af[i], wf[j], acc[i][j]);

#if !USE_ASYNC_LDS
        // stage after compute: staged values live load->store only (no spill);
        // the barrier below keeps loads from being hoisted across the WMMAs.
        asm volatile("" ::: "memory");
        if (more) stage(buf ^ 1, kk + 32);
#endif
        if (more) async_wait0();
        __syncthreads();
        buf ^= 1;
    }

#pragma unroll
    for (int j = 0; j < 4; j++) {
        int col = nt + wn * 64 + j * 16 + lm;
        float bv = bias ? bias[col] : 0.0f;
#pragma unroll
        for (int i = 0; i < 4; i++) {
#pragma unroll
            for (int v = 0; v < 8; v++) {
                int row = mt + wm * 64 + i * 16 + v + 8 * half;
                if (row < M) {
                    float val = acc[i][j][v] + bv;
                    if (flags & 1) val = fmaxf(val, 0.0f);
                    size_t off = (size_t)row * N + col;
                    if (flags & 2) Cf[off] = val;
                    if (flags & 4) Cb[off] = f2bf(val);
                }
            }
        }
    }
}

// ---------------------------------------------------------------------------
// Fused causal self-attention per (b,h): scores -> softmax -> P*V.
// Block: 128 threads (4 waves), 64 query rows per block, grid (S/64, B*H).
// Dynamic LDS: P (64x256 bf16, 32KB) + Vt (64x256 bf16, 32KB) = 64KB.
// QKV layout: row (b*S+s), 1536 cols = [q(512)|k(512)|v(512)], head h at h*64.
// Output: concat heads bf16 (B*S, 512).
// ---------------------------------------------------------------------------
__global__ void __launch_bounds__(128)
attn_fused(const unsigned short* __restrict__ QKV,
           unsigned short* __restrict__ OUT) {
    extern __shared__ unsigned short sm[];
    unsigned short* P  = sm;               // 64*256
    unsigned short* Vt = sm + 64 * 256;    // 64*256 (V transposed: [d][s])

    const int tid  = threadIdx.x;
    const int lane = tid & 31;
    const int wave = tid >> 5;
    const int lm   = lane & 15, half = lane >> 4;
    const int bh = blockIdx.y, b = bh >> 3, h = bh & 7;
    const int rb = blockIdx.x * 64;

    const unsigned short* Qg = QKV + (size_t)b * Sv * 1536 + h * 64;
    const unsigned short* Kg = Qg + 512;
    const unsigned short* Vg = Qg + 1024;

    // stage V^T into LDS (coalesced over d)
    for (int idx = tid; idx < 64 * 256; idx += 128) {
        int d = idx & 63, s2 = idx >> 6;
        Vt[d * 256 + s2] = Vg[(size_t)s2 * 1536 + d];
    }

    // ---- phase 1: scores (WMMA NT) + causal softmax ----
    const int q0 = rb + wave * 16;
    const unsigned short* qp = Qg + (size_t)(q0 + lm) * 1536 + half * 8;
    v16bf a0 = ld_frag(qp);        // K 0..31 of dh
    v16bf a1 = ld_frag(qp + 32);   // K 32..63

    v8f acc[16];
#pragma unroll
    for (int j = 0; j < 16; j++) {
        const unsigned short* kp = Kg + (size_t)(j * 16 + lm) * 1536 + half * 8;
        v8f c = (v8f){0,0,0,0,0,0,0,0};
        c = wmma_bf16(a0, ld_frag(kp), c);
        c = wmma_bf16(a1, ld_frag(kp + 32), c);
        acc[j] = c;
    }

#pragma unroll
    for (int v = 0; v < 8; v++) {
        int row = q0 + v + 8 * half;
        float sc[16];
        float mx = -3.0e38f;
#pragma unroll
        for (int j = 0; j < 16; j++) {
            int col = j * 16 + lm;
            float x = acc[j][v] * 0.125f;       // 1/sqrt(64)
            if (col > row) x += NEGV;           // causal mask
            sc[j] = x;
            mx = fmaxf(mx, x);
        }
#pragma unroll
        for (int o = 1; o < 16; o <<= 1) mx = fmaxf(mx, __shfl_xor(mx, o));
        float sum = 0.0f;
#pragma unroll
        for (int j = 0; j < 16; j++) { sc[j] = __expf(sc[j] - mx); sum += sc[j]; }
#pragma unroll
        for (int o = 1; o < 16; o <<= 1) sum += __shfl_xor(sum, o);
        float inv = 1.0f / sum;
        int rl = wave * 16 + v + 8 * half;
#pragma unroll
        for (int j = 0; j < 16; j++)
            P[rl * 256 + j * 16 + lm] = f2bf(sc[j] * inv);
    }
    __syncthreads();

    // ---- phase 2: O = P (64x256) x V (256x64), via Vt as NT operand ----
    v8f oacc[4];
#pragma unroll
    for (int j = 0; j < 4; j++) oacc[j] = (v8f){0,0,0,0,0,0,0,0};
#pragma unroll
    for (int kk = 0; kk < 256; kk += 32) {
        const unsigned short* pp = P + (wave * 16 + lm) * 256 + kk + half * 8;
        v16bf a = ld_frag(pp);
#pragma unroll
        for (int j = 0; j < 4; j++) {
            const unsigned short* vp = Vt + (j * 16 + lm) * 256 + kk + half * 8;
            oacc[j] = wmma_bf16(a, ld_frag(vp), oacc[j]);
        }
    }

    unsigned short* Ob = OUT + (size_t)b * Sv * Dv + h * 64;
#pragma unroll
    for (int j = 0; j < 4; j++)
#pragma unroll
        for (int v = 0; v < 8; v++) {
            int srow = rb + wave * 16 + v + 8 * half;
            Ob[(size_t)srow * Dv + j * 16 + lm] = f2bf(oacc[j][v]);
        }
}

// ---------------------------------------------------------------------------
// Residual add + LayerNorm. One wave per row (D=512, 16 elems/lane).
// delta row index = row / ddiv  (ddiv=1 normal, ddiv=S for broadcast cross-attn)
// Writes fp32 residual stream and bf16 copy.
// ---------------------------------------------------------------------------
__global__ void __launch_bounds__(256)
ln_kernel(const float* __restrict__ X, const float* __restrict__ Dl,
          const float* __restrict__ sc, const float* __restrict__ bt,
          float* __restrict__ Xo, unsigned short* __restrict__ Xb,
          int rows, int ddiv) {
    const int lane = threadIdx.x & 31;
    const int row  = blockIdx.x * 8 + (threadIdx.x >> 5);
    if (row >= rows) return;
    const float* xr = X  + (size_t)row * Dv;
    const float* dr = Dl + (size_t)(row / ddiv) * Dv;

    float r[16];
    float sum = 0.0f;
#pragma unroll
    for (int i = 0; i < 16; i++) {
        int c = lane + i * 32;
        r[i] = xr[c] + dr[c];
        sum += r[i];
    }
#pragma unroll
    for (int o = 1; o < 32; o <<= 1) sum += __shfl_xor(sum, o);
    float mean = sum * (1.0f / Dv);
    float vs = 0.0f;
#pragma unroll
    for (int i = 0; i < 16; i++) { float d = r[i] - mean; vs += d * d; }
#pragma unroll
    for (int o = 1; o < 32; o <<= 1) vs += __shfl_xor(vs, o);
    float inv = rsqrtf(vs * (1.0f / Dv) + 1e-5f);
#pragma unroll
    for (int i = 0; i < 16; i++) {
        int c = lane + i * 32;
        float y = (r[i] - mean) * inv * sc[c] + bt[c];
        Xo[(size_t)row * Dv + c] = y;
        Xb[(size_t)row * Dv + c] = f2bf(y);
    }
}

// ---------------------------------------------------------------------------
// Host launcher
// ---------------------------------------------------------------------------
extern "C" void kernel_launch(void* const* d_in, const int* in_sizes, int n_in,
                              void* d_out, int out_size, void* d_ws, size_t ws_size,
                              hipStream_t stream) {
    (void)in_sizes; (void)n_in; (void)out_size; (void)ws_size;

    const float* z      = (const float*)d_in[0];
    const int*   tgt    = (const int*)  d_in[1];
    const float* emb    = (const float*)d_in[2];
    const float* pos    = (const float*)d_in[3];
    const float* proj_w = (const float*)d_in[4];
    const float* proj_b = (const float*)d_in[5];
    const float* sa_w   = (const float*)d_in[6];
    const float* sa_b   = (const float*)d_in[7];
    const float* sa_ow  = (const float*)d_in[8];
    const float* sa_ob  = (const float*)d_in[9];
    const float* ca_w   = (const float*)d_in[10];
    const float* ca_b   = (const float*)d_in[11];
    const float* ca_ow  = (const float*)d_in[12];
    const float* ca_ob  = (const float*)d_in[13];
    const float* ln1_s  = (const float*)d_in[14];
    const float* ln1_b  = (const float*)d_in[15];
    const float* ln2_s  = (const float*)d_in[16];
    const float* ln2_b  = (const float*)d_in[17];
    const float* ln3_s  = (const float*)d_in[18];
    const float* ln3_b  = (const float*)d_in[19];
    const float* ff1_w  = (const float*)d_in[20];
    const float* ff1_b  = (const float*)d_in[21];
    const float* ff2_w  = (const float*)d_in[22];
    const float* ff2_b  = (const float*)d_in[23];
    const float* fc_w   = (const float*)d_in[24];
    const float* fc_b   = (const float*)d_in[25];

    // ---- workspace carve-up ----
    char* base = (char*)d_ws;
    size_t off = 0;
    auto take = [&](size_t bytes) -> void* {
        void* p = base + off;
        off = (off + bytes + 255) & ~(size_t)255;
        return p;
    };
    const size_t E_PROJ = 512 * 512,  E_SAW = 6 * 1536 * 512,
                 E_SAOW = 6 * 512 * 512, E_CAW = 6 * 1536 * 512,
                 E_CAOW = 6 * 512 * 512, E_FF1 = 6 * 2048 * 512,
                 E_FF2 = 6 * 512 * 2048, E_FC = 128 * 512, E_Z = 64 * 512;

    unsigned short* WBproj = (unsigned short*)take(E_PROJ * 2);
    unsigned short* WBsaw  = (unsigned short*)take(E_SAW  * 2);
    unsigned short* WBsaow = (unsigned short*)take(E_SAOW * 2);
    unsigned short* WBcaw  = (unsigned short*)take(E_CAW  * 2);
    unsigned short* WBcaow = (unsigned short*)take(E_CAOW * 2);
    unsigned short* WBff1  = (unsigned short*)take(E_FF1  * 2);
    unsigned short* WBff2  = (unsigned short*)take(E_FF2  * 2);
    unsigned short* WBfc   = (unsigned short*)take(E_FC   * 2);
    unsigned short* ZB     = (unsigned short*)take(E_Z    * 2);

    float*          XF     = (float*)         take((size_t)Bv * Sv * Dv * 4);
    unsigned short* XB     = (unsigned short*)take((size_t)Bv * Sv * Dv * 2);
    float*          TMP    = (float*)         take((size_t)Bv * Sv * Dv * 4);
    unsigned short* BIG    = (unsigned short*)take((size_t)Bv * Sv * DFFv * 2); // qkv / ffn hidden (aliased)
    unsigned short* CONCAT = (unsigned short*)take((size_t)Bv * Sv * Dv * 2);
    unsigned short* MEMB   = (unsigned short*)take((size_t)Bv * Dv * 2);
    unsigned short* VCB    = (unsigned short*)take((size_t)Bv * Dv * 2);
    float*          CTMP   = (float*)         take((size_t)Bv * Dv * 4);

    auto cvt = [&](const float* s, unsigned short* d, size_t n) {
        cvt_bf16<<<2048, 256, 0, stream>>>(s, d, (int)n);
    };

    // ---- weight conversion (once per launch; deterministic) ----
    cvt(proj_w, WBproj, E_PROJ);
    cvt(sa_w,   WBsaw,  E_SAW);
    cvt(sa_ow,  WBsaow, E_SAOW);
    cvt(ca_w,   WBcaw,  E_CAW);
    cvt(ca_ow,  WBcaow, E_CAOW);
    cvt(ff1_w,  WBff1,  E_FF1);
    cvt(ff2_w,  WBff2,  E_FF2);
    cvt(fc_w,   WBfc,   E_FC);
    cvt(z,      ZB,     E_Z);

    // ---- embedding + positional ----
    embed_kernel<<<8192, 256, 0, stream>>>(tgt, emb, pos, XF, XB);

    // ---- memory projection: mem = z @ proj_w.T + proj_b  (64x512) ----
    gemm_nt<<<dim3(4, 1), 128, 0, stream>>>(ZB, WBproj, proj_b,
                                            nullptr, MEMB, 64, 512, 512, 4);

    const int Mtok = Bv * Sv; // 16384

    for (int i = 0; i < Lv; i++) {
        // self-attn QKV: (16384 x 1536)
        gemm_nt<<<dim3(12, 128), 128, 0, stream>>>(
            XB, WBsaw + (size_t)i * 1536 * 512, sa_b + (size_t)i * 1536,
            nullptr, BIG, Mtok, 1536, 512, 4);

        // fused causal attention -> CONCAT (bf16, 16384 x 512)
        attn_fused<<<dim3(4, Bv * Hv), 128, 65536, stream>>>(BIG, CONCAT);

        // out projection -> TMP (f32)
        gemm_nt<<<dim3(4, 128), 128, 0, stream>>>(
            CONCAT, WBsaow + (size_t)i * 512 * 512, sa_ob + (size_t)i * 512,
            TMP, nullptr, Mtok, 512, 512, 2);

        // x = LN(x + sa)
        ln_kernel<<<2048, 256, 0, stream>>>(XF, TMP,
            ln1_s + (size_t)i * 512, ln1_b + (size_t)i * 512,
            XF, XB, Mtok, 1);

        // cross-attn (1 key -> softmax==1): vc = mem @ wv.T + bv
        gemm_nt<<<dim3(4, 1), 128, 0, stream>>>(
            MEMB, WBcaw + (size_t)i * 1536 * 512 + (size_t)1024 * 512,
            ca_b + (size_t)i * 1536 + 1024,
            nullptr, VCB, 64, 512, 512, 4);

        // cout = vc @ ow.T + ob  (64x512, broadcast over s)
        gemm_nt<<<dim3(4, 1), 128, 0, stream>>>(
            VCB, WBcaow + (size_t)i * 512 * 512, ca_ob + (size_t)i * 512,
            CTMP, nullptr, 64, 512, 512, 2);

        // x = LN(x + cout[b])  (broadcast: delta row = row / S)
        ln_kernel<<<2048, 256, 0, stream>>>(XF, CTMP,
            ln2_s + (size_t)i * 512, ln2_b + (size_t)i * 512,
            XF, XB, Mtok, Sv);

        // FFN: h = relu(x @ ff1.T + b1)  (16384 x 2048, bf16)
        gemm_nt<<<dim3(16, 128), 128, 0, stream>>>(
            XB, WBff1 + (size_t)i * 2048 * 512, ff1_b + (size_t)i * 2048,
            nullptr, BIG, Mtok, 2048, 512, 4 | 1);

        // delta = h @ ff2.T + b2  (16384 x 512, f32)
        gemm_nt<<<dim3(4, 128), 128, 0, stream>>>(
            BIG, WBff2 + (size_t)i * 512 * 2048, ff2_b + (size_t)i * 512,
            TMP, nullptr, Mtok, 512, 2048, 2);

        // x = LN(x + ff)
        ln_kernel<<<2048, 256, 0, stream>>>(XF, TMP,
            ln3_s + (size_t)i * 512, ln3_b + (size_t)i * 512,
            XF, XB, Mtok, 1);
    }

    // ---- logits: x @ fc_w.T + fc_b  (16384 x 128) -> d_out (f32) ----
    gemm_nt<<<dim3(1, 128), 128, 0, stream>>>(
        XB, WBfc, fc_b, (float*)d_out, nullptr, Mtok, 128, 512, 2);
}
